// LSTMDirectionModel_8083128451753
// MI455X (gfx1250) — compile-verified, matrix operand order checked by reading, not run on hardware
//
#include <hip/hip_runtime.h>
#include <hip/hip_bf16.h>
#include <math.h>

typedef __attribute__((ext_vector_type(16))) __bf16 bf16x16;
typedef __attribute__((ext_vector_type(8)))  __bf16 bf16x8;
typedef __attribute__((ext_vector_type(8)))  float  f32x8;

#define BB 256      // batch
#define TT 512      // seq len
#define DD 64       // input dim (layer 0)
#define HH 256      // hidden
#define G4H 1024    // 4*H gates
#define MB (BB/16)  // 16 batch blocks of 16 rows

// LDS: h tile (16x256) + gates g,o of Whh (32 N-tiles x 8 K-tiles x 512)
#define LDS_H_ELEMS   (16 * HH)
#define LDS_W_ELEMS   (32 * 8 * 512)
#define LDS_BYTES     ((LDS_H_ELEMS + LDS_W_ELEMS) * 2)

// Pair-offset within a 16-bit WMMA A/B operand row (half*8 folded into base).
__device__ __forceinline__ int kpair(int j) { return ((j & 3) << 1) + ((j >> 2) << 4); }

__device__ __forceinline__ float sigm_(float x) { return 1.0f / (1.0f + __expf(-x)); }
__device__ __forceinline__ float tanh_(float x) {
  x = fminf(fmaxf(x, -15.0f), 15.0f);
  float e = __expf(2.0f * x);
  return (e - 1.0f) / (e + 1.0f);
}

// x f32 [B,T,D] -> xs0 bf16 [T,B,D]
__global__ void convert_x_kernel(const float* __restrict__ x, __bf16* __restrict__ xs0) {
  int idx = blockIdx.x * blockDim.x + threadIdx.x;
  if (idx >= TT * BB * DD) return;
  int d = idx % DD;
  int b = (idx / DD) % BB;
  int t = idx / (DD * BB);
  xs0[idx] = (__bf16)x[((size_t)b * TT + t) * DD + d];
}

// W f32 [N,K] row-major -> bf16 WMMA-B tiles, per-lane 32B contiguous.
__global__ void swizzle_w_kernel(const float* __restrict__ W, __bf16* __restrict__ out,
                                 int N, int K) {
  int idx = blockIdx.x * blockDim.x + threadIdx.x;
  if (idx >= N * K) return;
  int e    = idx & 1;
  int j    = (idx >> 1) & 7;
  int lane = (idx >> 4) & 31;
  int tile = idx >> 9;
  int nKt  = K >> 5;
  int kt = tile % nKt, nt = tile / nKt;
  int n = lane & 15, half = lane >> 4;
  int k = kpair(j) + (half << 3) + e;
  out[idx] = (__bf16)W[(size_t)(nt * 16 + n) * K + (kt * 32 + k)];
}

// gx = xs @ Wih^T + (bih + bhh). 512 threads; wave owns 4 of the 64 N-tiles.
template <int NKT>
__global__ void __launch_bounds__(512) gemm_gx_kernel(
    const __bf16* __restrict__ xs, const __bf16* __restrict__ Wsw,
    const float* __restrict__ bih, const float* __restrict__ bhh,
    __bf16* __restrict__ gx) {
  int mt   = blockIdx.x;            // t*MB + b_blk
  int lane = threadIdx.x & 31;
  int wid  = threadIdx.x >> 5;      // 0..15
  int half = lane >> 4;
  int mrow = lane & 15;
  const __bf16* arow = xs + ((size_t)mt * 16 + mrow) * (NKT * 32) + (half << 3);
  const __bf16* wb   = Wsw + (size_t)wid * 4 * NKT * 512 + lane * 16;
  f32x8 acc[4] = {};
  #pragma unroll 2
  for (int kt = 0; kt < NKT; ++kt) {
    bf16x16 a;
    #pragma unroll
    for (int j = 0; j < 8; ++j) {
      int k = kpair(j);
      a[2 * j] = arow[k]; a[2 * j + 1] = arow[k + 1];
    }
    bf16x16 b[4];
    #pragma unroll
    for (int i = 0; i < 4; ++i)
      b[i] = *(const bf16x16*)(wb + (size_t)i * NKT * 512);
    #pragma unroll
    for (int i = 0; i < 4; ++i)
      acc[i] = __builtin_amdgcn_wmma_f32_16x16x32_bf16(false, a, false, b[i],
                                                       (short)0, acc[i], false, false);
    arow += 32;
    wb   += 512;
  }
  __bf16* gxo = gx + ((size_t)mt * 64 + wid * 4) * 256 + lane * 8;
  #pragma unroll
  for (int i = 0; i < 4; ++i) {
    int n = (wid * 4 + i) * 16 + (lane & 15);
    float bsum = bih[n] + bhh[n];
    bf16x8 o;
    #pragma unroll
    for (int r = 0; r < 8; ++r) o[r] = (__bf16)(acc[i][r] + bsum);
    *(bf16x8*)(gxo + i * 256) = o;
  }
}

// Recurrence, both directions concurrently: 32 WGs x 512 threads (16 waves).
// Wave w owns h-column block jblk == w for all 4 gates. Whh split across the
// two big CDNA5 pools: gates i,f pinned in VGPRs (128 regs/wave), gates g,o
// staged once in LDS (256 KB). Explicit 1-deep double buffer; a
// sched_barrier(0) at the end of each k-iteration pins the pipeline depth
// (prevents both the round-4 rotating-register v_mov storm and the round-5
// volatile->flat/SYS serialization).
__global__ void __launch_bounds__(512) lstm_both_kernel(
    const __bf16* __restrict__ gxF, const __bf16* __restrict__ gxB,
    const __bf16* __restrict__ WhhF, const __bf16* __restrict__ WhhB,
    __bf16* __restrict__ xs_out) {
  extern __shared__ __bf16 smem[];
  __bf16* hsh = smem;                  // [16][HH]
  __bf16* wsh = smem + LDS_H_ELEMS;    // gates g,o swizzled tiles (contiguous)

  int dir  = blockIdx.x >> 4;
  int bblk = blockIdx.x & 15;
  int lane = threadIdx.x & 31;
  int wid  = threadIdx.x >> 5;      // 0..15 == jblk
  int half = lane >> 4;
  int mrow = lane & 15;

  const __bf16* gx  = dir ? gxB : gxF;
  const __bf16* Whh = dir ? WhhB : WhhF;
  int dirOff = dir * HH;
  int t0 = dir ? (TT - 1) : 0;
  ptrdiff_t tsgn = dir ? -1 : 1;

  // Pin gates 0,1 (i,f) weights in VGPRs: tile (nt = g*16 + wid, kt).
  const __bf16* wbg = Whh + (size_t)wid * 8 * 512 + lane * 16;
  bf16x16 breg[16];
  #pragma unroll
  for (int g = 0; g < 2; ++g)
    #pragma unroll
    for (int kt = 0; kt < 8; ++kt)
      breg[g * 8 + kt] = *(const bf16x16*)(wbg + (size_t)(g * 128 + kt) * 512);

  // Stage gates 2,3 (g,o) weights into LDS (straight copy of swizzled region
  // starting at tile nt=32: element offset 32*8*512 = 131072).
  {
    const __bf16* src = Whh + 32 * 8 * 512;
    for (int i = threadIdx.x * 8; i < LDS_W_ELEMS; i += 512 * 8)
      *(bf16x8*)(wsh + i) = *(const bf16x8*)(src + i);
  }
  // Per-wave LDS bases for the two staged gates (128KB apart > 16-bit ds imm).
  const __bf16* wl2 = wsh + (size_t)wid * 8 * 512 + lane * 16;            // gate 2
  const __bf16* wl3 = wl2 + 16 * 8 * 512;                                 // gate 3

  const __bf16* arow = &hsh[mrow * HH + (half << 3)];
  const __bf16* gxp  = gx + (((size_t)t0 * MB + bblk) * 64 + (size_t)wid) * 256 + lane * 8;
  ptrdiff_t gxstride = tsgn * (ptrdiff_t)MB * 64 * 256;
  __bf16* xsp = xs_out + ((size_t)t0 * BB + bblk * 16) * (2 * HH) + dirOff;
  ptrdiff_t xsstride = tsgn * (ptrdiff_t)BB * (2 * HH);
  // cooperative h write-out: 512 threads x 8 cols (one b128 each)
  int cm = threadIdx.x >> 5;          // row 0..15
  int cc = (threadIdx.x & 31) * 8;    // 8-col chunk

  float c[8] = {};
  for (int i = threadIdx.x; i < LDS_H_ELEMS; i += 512) hsh[i] = (__bf16)0.0f;
  __syncthreads();   // h zeroed + weights staged

  for (int s = 0; s < TT; ++s) {
    // z += h @ Whh^T, explicit 1-deep double buffer (even/odd register sets).
    bf16x16 a[2], b2[2], b3[2];
    #pragma unroll
    for (int j = 0; j < 8; ++j) {
      int k = kpair(j);
      a[0][2 * j] = arow[k]; a[0][2 * j + 1] = arow[k + 1];
    }
    b2[0] = *(const bf16x16*)(wl2);
    b3[0] = *(const bf16x16*)(wl3);
    f32x8 acc[4] = {};
    #pragma unroll
    for (int kt = 0; kt < 8; ++kt) {
      int cur = kt & 1, nx = cur ^ 1;
      if (kt < 7) {
        #pragma unroll
        for (int j = 0; j < 8; ++j) {
          int k = (kt + 1) * 32 + kpair(j);
          a[nx][2 * j] = arow[k]; a[nx][2 * j + 1] = arow[k + 1];
        }
        b2[nx] = *(const bf16x16*)(wl2 + (size_t)(kt + 1) * 512);
        b3[nx] = *(const bf16x16*)(wl3 + (size_t)(kt + 1) * 512);
      }
      acc[0] = __builtin_amdgcn_wmma_f32_16x16x32_bf16(false, a[cur], false, breg[kt],
                                                       (short)0, acc[0], false, false);
      acc[1] = __builtin_amdgcn_wmma_f32_16x16x32_bf16(false, a[cur], false, breg[8 + kt],
                                                       (short)0, acc[1], false, false);
      acc[2] = __builtin_amdgcn_wmma_f32_16x16x32_bf16(false, a[cur], false, b2[cur],
                                                       (short)0, acc[2], false, false);
      acc[3] = __builtin_amdgcn_wmma_f32_16x16x32_bf16(false, a[cur], false, b3[cur],
                                                       (short)0, acc[3], false, false);
      __builtin_amdgcn_sched_barrier(0);   // pin pipeline depth to 1 iteration
    }

    // Gate-input tiles: global loads, latency covered by the barrier below.
    bf16x8 gv[4];
    #pragma unroll
    for (int g = 0; g < 4; ++g)
      gv[g] = *(const bf16x8*)(gxp + (size_t)g * 4096);
    __syncthreads();   // all waves done reading h before overwrite

    int jcol = wid * 16 + (lane & 15);
    #pragma unroll
    for (int r = 0; r < 8; ++r) {
      float zi = acc[0][r] + (float)gv[0][r];
      float zf = acc[1][r] + (float)gv[1][r];
      float zg = acc[2][r] + (float)gv[2][r];
      float zo = acc[3][r] + (float)gv[3][r];
      float cn = sigm_(zf) * c[r] + sigm_(zi) * tanh_(zg);
      c[r] = cn;
      float hv = sigm_(zo) * tanh_(cn);
      int m = r + (half << 3);
      hsh[m * HH + jcol] = (__bf16)hv;
    }
    __syncthreads();   // h fully updated

    // Coalesced write-out of this step's h; overlaps next step's WMMAs.
    *(bf16x8*)(xsp + (size_t)cm * (2 * HH) + cc) = *(const bf16x8*)(&hsh[cm * HH + cc]);
    gxp += gxstride;
    xsp += xsstride;
  }
}

// h1 = relu(concat(hT_f, hT_b) @ fc1_w^T + b)
__global__ void __launch_bounds__(256) fc1_kernel(
    const __bf16* __restrict__ xsF, const __bf16* __restrict__ Wsw,
    const float* __restrict__ bias, __bf16* __restrict__ h1) {
  int mtile = blockIdx.x;
  int lane = threadIdx.x & 31;
  int wid  = threadIdx.x >> 5;
  int half = lane >> 4;
  int mrow = lane & 15;
  const __bf16* rowF = xsF + ((size_t)(TT - 1) * BB + mtile * 16 + mrow) * (2 * HH) + (half << 3);
  const __bf16* rowB = xsF + ((size_t)(mtile * 16 + mrow)) * (2 * HH) + (half << 3);
  const __bf16* wb = Wsw + (size_t)(wid * 2) * 16 * 512 + lane * 16;
  f32x8 acc[2] = {};
  #pragma unroll 4
  for (int kt = 0; kt < 16; ++kt) {
    const __bf16* arow = (kt < 8) ? rowF : rowB;
    bf16x16 a;
    #pragma unroll
    for (int j = 0; j < 8; ++j) {
      int k = kt * 32 + kpair(j);
      a[2 * j] = arow[k]; a[2 * j + 1] = arow[k + 1];
    }
    #pragma unroll
    for (int i = 0; i < 2; ++i) {
      bf16x16 b = *(const bf16x16*)(wb + (size_t)(i * 16 + kt) * 512);
      acc[i] = __builtin_amdgcn_wmma_f32_16x16x32_bf16(false, a, false, b,
                                                       (short)0, acc[i], false, false);
    }
  }
  #pragma unroll
  for (int i = 0; i < 2; ++i) {
    int n = (wid * 2 + i) * 16 + (lane & 15);
    float bv = bias[n];
    #pragma unroll
    for (int r = 0; r < 8; ++r) {
      float v = acc[i][r] + bv;
      v = v > 0.0f ? v : 0.0f;
      int m = r + (half << 3);
      h1[(size_t)(mtile * 16 + m) * HH + n] = (__bf16)v;
    }
  }
}

__global__ void fc2_kernel(const __bf16* __restrict__ h1, const float* __restrict__ w2,
                           const float* __restrict__ b2, float* __restrict__ out) {
  int b = blockIdx.x * blockDim.x + threadIdx.x;
  if (b >= BB) return;
  float s = b2[0];
  for (int j = 0; j < HH; ++j) s += (float)h1[(size_t)b * HH + j] * w2[j];
  out[b] = 1.0f / (1.0f + __expf(-s));
}

extern "C" void kernel_launch(void* const* d_in, const int* in_sizes, int n_in,
                              void* d_out, int out_size, void* d_ws, size_t ws_size,
                              hipStream_t stream) {
  (void)in_sizes; (void)n_in; (void)out_size; (void)ws_size;
  const float* x    = (const float*)d_in[0];
  const float* Wih0 = (const float*)d_in[1];
  const float* Whh0 = (const float*)d_in[2];
  const float* bih0 = (const float*)d_in[3];
  const float* bhh0 = (const float*)d_in[4];
  const float* WihL = (const float*)d_in[5];
  const float* WhhL = (const float*)d_in[6];
  const float* bihL = (const float*)d_in[7];
  const float* bhhL = (const float*)d_in[8];
  const float* fc1w = (const float*)d_in[9];
  const float* fc1b = (const float*)d_in[10];
  const float* fc2w = (const float*)d_in[11];
  const float* fc2b = (const float*)d_in[12];
  float* out = (float*)d_out;

  // Workspace layout (bf16 elements)
  __bf16* w = (__bf16*)d_ws;
  const size_t XS  = (size_t)TT * BB * (2 * HH);
  const size_t GXS = (size_t)TT * MB * 64 * 256;
  __bf16* xsA = w;
  __bf16* xsB = xsA + XS;
  __bf16* gxF = xsB + XS;
  __bf16* gxB = gxF + GXS;
  __bf16* p   = gxB + GXS;
  __bf16* wWih0[2];    for (int d = 0; d < 2; ++d) { wWih0[d] = p; p += (size_t)G4H * DD; }
  __bf16* wWhh0[2];    for (int d = 0; d < 2; ++d) { wWhh0[d] = p; p += (size_t)G4H * HH; }
  __bf16* wWihX[2][2]; for (int l = 0; l < 2; ++l) for (int d = 0; d < 2; ++d) { wWihX[l][d] = p; p += (size_t)G4H * 2 * HH; }
  __bf16* wWhhX[2][2]; for (int l = 0; l < 2; ++l) for (int d = 0; d < 2; ++d) { wWhhX[l][d] = p; p += (size_t)G4H * HH; }
  __bf16* wFc1 = p; p += (size_t)HH * 2 * HH;
  __bf16* h1   = p; p += (size_t)BB * HH;

  convert_x_kernel<<<(TT * BB * DD + 255) / 256, 256, 0, stream>>>(x, xsA);
  auto sw = [&](const float* src, __bf16* dst, int N, int K) {
    swizzle_w_kernel<<<(N * K + 255) / 256, 256, 0, stream>>>(src, dst, N, K);
  };
  for (int d = 0; d < 2; ++d) {
    sw(Wih0 + (size_t)d * G4H * DD, wWih0[d], G4H, DD);
    sw(Whh0 + (size_t)d * G4H * HH, wWhh0[d], G4H, HH);
  }
  for (int l = 0; l < 2; ++l)
    for (int d = 0; d < 2; ++d) {
      sw(WihL + (size_t)(l * 2 + d) * G4H * 2 * HH, wWihX[l][d], G4H, 2 * HH);
      sw(WhhL + (size_t)(l * 2 + d) * G4H * HH,     wWhhX[l][d], G4H, HH);
    }
  sw(fc1w, wFc1, HH, 2 * HH);

  // Layer 0 (K = 64)
  gemm_gx_kernel<2><<<TT * MB, 512, 0, stream>>>(xsA, wWih0[0], bih0, bhh0, gxF);
  gemm_gx_kernel<2><<<TT * MB, 512, 0, stream>>>(xsA, wWih0[1], bih0 + G4H, bhh0 + G4H, gxB);
  lstm_both_kernel<<<2 * MB, 512, LDS_BYTES, stream>>>(gxF, gxB, wWhh0[0], wWhh0[1], xsB);

  // Layers 1..2 (K = 512)
  __bf16* cur = xsB; __bf16* nxt = xsA;
  for (int l = 0; l < 2; ++l) {
    gemm_gx_kernel<16><<<TT * MB, 512, 0, stream>>>(cur, wWihX[l][0],
        bihL + (size_t)(l * 2 + 0) * G4H, bhhL + (size_t)(l * 2 + 0) * G4H, gxF);
    gemm_gx_kernel<16><<<TT * MB, 512, 0, stream>>>(cur, wWihX[l][1],
        bihL + (size_t)(l * 2 + 1) * G4H, bhhL + (size_t)(l * 2 + 1) * G4H, gxB);
    lstm_both_kernel<<<2 * MB, 512, LDS_BYTES, stream>>>(gxF, gxB, wWhhX[l][0], wWhhX[l][1], nxt);
    __bf16* t = cur; cur = nxt; nxt = t;
  }

  fc1_kernel<<<MB, 256, 0, stream>>>(cur, wFc1, fc1b, h1);
  fc2_kernel<<<1, 256, 0, stream>>>(h1, fc2w, fc2b, out);
}
